// BinaryLenet_38019050504778
// MI455X (gfx1250) — 8-bit matrix operand order repaired and checked by reading, compiled, not run on hardware
//
#include <hip/hip_runtime.h>
#include <cstdint>
#include <cstddef>


typedef __attribute__((ext_vector_type(16))) int   v16i;
typedef __attribute__((ext_vector_type(8)))  float v8f;

#define NB       4096
#define NC2      50
#define FCH      1024
#define K2       500
#define K2P      512
#define K3       2450
#define K3P      2560
#define H1P_IMG  (18*18*20)

#define F8_P1 0x38
#define F8_M1 0xB8

__device__ __forceinline__ int kofsA(int v, int h) {
  return ((v >> 2) << 5) | (h << 4) | ((v & 3) << 2);
}
__device__ __forceinline__ int kofsB(int v, int h) {
  return ((v >> 2) << 5) | (h << 4) | ((v & 3) << 2);
}


__global__ void k_prep_w2(const float* __restrict__ w2, uint8_t* __restrict__ W2b) {
  int id = blockIdx.x * blockDim.x + threadIdx.x;
  if (id >= 64 * K2P) return;
  int n = id >> 9, k = id & 511;
  uint8_t o = 0;
  if (n < NC2 && k < K2) {
    int dy = k / 100, r = k - dy * 100;
    int dx = r / 20,  c = r - dx * 20;
    float w = w2[((n * 20 + c) * 5 + dy) * 5 + dx];
    o = (w > 0.f) ? F8_P1 : ((w < 0.f) ? F8_M1 : 0);
  }
  W2b[id] = o;
}

__global__ void k_prep_w3(const float* __restrict__ w3, uint8_t* __restrict__ W3b) {
  int id = blockIdx.x * blockDim.x + threadIdx.x;
  if (id >= FCH * K3P) return;
  int n = id / K3P, k = id - n * K3P;
  uint8_t o = 0;
  if (k < K3) {
    float w = w3[(size_t)n * K3 + k];
    o = (w > 0.f) ? F8_P1 : ((w < 0.f) ? F8_M1 : 0);
  }
  W3b[id] = o;
}

__global__ void k_prep_w4(const float* __restrict__ w4, uint8_t* __restrict__ W4b) {
  int id = blockIdx.x * blockDim.x + threadIdx.x;
  if (id >= 16 * FCH) return;
  int n = id >> 10, k = id & 1023;
  uint8_t o = 0;
  if (n < 10) {
    float w = w4[n * FCH + k];
    o = (w > 0.f) ? F8_P1 : ((w < 0.f) ? F8_M1 : 0);
  }
  W4b[id] = o;
}

__global__ void k_zero32(uint32_t* __restrict__ p, int n) {
  int id = blockIdx.x * blockDim.x + threadIdx.x;
  if (id < n) p[id] = 0u;
}


__global__ void k_conv1(const float* __restrict__ x, const float* __restrict__ w1,
                        const float* __restrict__ b1, uint8_t* __restrict__ h1p) {
  int id = blockIdx.x * blockDim.x + threadIdx.x;
  if (id >= NB * 324) return;
  int b = id / 324, r = id - b * 324;
  int py = r / 18, px = r - py * 18;
  uint8_t* o = h1p + (size_t)id * 20;
  if (py < 2 || py >= 16 || px < 2 || px >= 16) {
    #pragma unroll
    for (int c = 0; c < 20; c++) o[c] = 0;
    return;
  }
  int y = py - 2, xp = px - 2;
  int gy0 = 2 * y - 2, gx0 = 2 * xp - 2;
  const float* xb = x + (size_t)b * 784;
  int sx[6][6];
  #pragma unroll
  for (int i = 0; i < 6; i++) {
    int gy = gy0 + i;
    #pragma unroll
    for (int j = 0; j < 6; j++) {
      int gx = gx0 + j, s = 0;
      if (gy >= 0 && gy < 28 && gx >= 0 && gx < 28) {
        float v = xb[gy * 28 + gx];
        s = (v > 0.f) ? 1 : ((v < 0.f) ? -1 : 0);
      }
      sx[i][j] = s;
    }
  }
  for (int c = 0; c < 20; c++) {
    int s00 = 0, s01 = 0, s10 = 0, s11 = 0;
    #pragma unroll
    for (int i = 0; i < 5; i++)
      #pragma unroll
      for (int j = 0; j < 5; j++) {
        float w = w1[c * 25 + i * 5 + j];
        int sw = (w > 0.f) ? 1 : -1;
        s00 += sx[i][j]     * sw;  s01 += sx[i][j + 1]     * sw;
        s10 += sx[i + 1][j] * sw;  s11 += sx[i + 1][j + 1] * sw;
      }
    int mx = max(max(s00, s01), max(s10, s11));
    float v = (float)mx + b1[c];
    o[c] = (v > 0.f) ? F8_P1 : 0;
  }
}


__global__ void __launch_bounds__(256)
k_conv2(const uint8_t* __restrict__ h1p, const uint8_t* __restrict__ W2b,
        const float* __restrict__ b2, uint8_t* __restrict__ C2bits) {
  int lane = threadIdx.x & 31, wid = threadIdx.x >> 5;
  int half = lane >> 4, lm = lane & 15;
  int mt = blockIdx.x * 8 + wid;
  int m0 = mt << 4;
  int mrow = m0 + lm;
  int b = mrow / 196, pix = mrow - b * 196;
  int y = pix / 14, xp = pix - y * 14;
  const uint8_t* abase = h1p + (size_t)b * H1P_IMG + y * 360 + xp * 20;

  v8f zero = {0.f, 0.f, 0.f, 0.f, 0.f, 0.f, 0.f, 0.f};
  v8f acc[4] = {zero, zero, zero, zero};

  #pragma unroll
  for (int kc = 0; kc < 4; kc++) {
    v16i a;
    #pragma unroll
    for (int v = 0; v < 16; v++) {
      int k = kc * 128 + kofsA(v, half);
      int dw = 0;
      if (k < K2) {
        int dy = k / 100, off = k - dy * 100;
        dw = *(const int*)(abase + dy * 360 + off);
      }
      a[v] = dw;
    }
    #pragma unroll
    for (int t = 0; t < 4; t++) {
      const uint8_t* bbase = W2b + (t * 16 + lm) * K2P + kc * 128;
      v16i bm;
      #pragma unroll
      for (int v = 0; v < 16; v++)
        bm[v] = *(const int*)(bbase + kofsB(v, half));
      acc[t] = __builtin_amdgcn_wmma_f32_16x16x128_fp8_fp8(
                   a, bm, (short)0, acc[t], false, false);
    }
  }
  #pragma unroll
  for (int t = 0; t < 4; t++) {
    int n = t * 16 + lm;
    if (n < NC2) {
      float bias = b2[n];
      #pragma unroll
      for (int r = 0; r < 8; r++) {
        int m = m0 + (half << 3) + r;
        float v = acc[t][r] + bias;
        C2bits[(size_t)m * NC2 + n] = (v > 0.f) ? 1 : 0;
      }
    }
  }
}

__global__ void k_pool2(const uint8_t* __restrict__ C2bits, uint8_t* __restrict__ A1) {
  int id = blockIdx.x * blockDim.x + threadIdx.x;
  if (id >= NB * 49) return;
  int b = id / 49, r = id - b * 49;
  int y = r / 7, xp = r - y * 7;
  const uint8_t* p0 = C2bits + (size_t)(b * 196 + 2 * y * 14 + 2 * xp) * NC2;
  uint8_t* o = A1 + (size_t)b * K3P;
  for (int c = 0; c < NC2; c++) {
    uint8_t v = p0[c] | p0[NC2 + c] | p0[14 * NC2 + c] | p0[15 * NC2 + c];
    o[c * 49 + y * 7 + xp] = v ? F8_P1 : 0;
  }
}

__global__ void __launch_bounds__(256)
k_fc1(const uint8_t* __restrict__ A1, const uint8_t* __restrict__ W3b,
      const float* __restrict__ b3, uint8_t* __restrict__ H2) {
  int lane = threadIdx.x & 31, wid = threadIdx.x >> 5;
  int half = lane >> 4, lm = lane & 15;
  int wg = blockIdx.x * 8 + wid;
  int mt = wg >> 4, ng = wg & 15;
  int m0 = mt << 4, n0 = ng << 6;
  const uint8_t* arow = A1 + (size_t)(m0 + lm) * K3P;

  v8f zero = {0.f, 0.f, 0.f, 0.f, 0.f, 0.f, 0.f, 0.f};
  v8f acc[4] = {zero, zero, zero, zero};

  for (int kc = 0; kc < 20; kc++) {
    v16i a;
    #pragma unroll
    for (int v = 0; v < 16; v++)
      a[v] = *(const int*)(arow + kc * 128 + kofsA(v, half));
    #pragma unroll
    for (int t = 0; t < 4; t++) {
      const uint8_t* brow = W3b + (size_t)(n0 + t * 16 + lm) * K3P + kc * 128;
      v16i bm;
      #pragma unroll
      for (int v = 0; v < 16; v++)
        bm[v] = *(const int*)(brow + kofsB(v, half));
      acc[t] = __builtin_amdgcn_wmma_f32_16x16x128_fp8_fp8(
                   a, bm, (short)0, acc[t], false, false);
    }
  }
  #pragma unroll
  for (int t = 0; t < 4; t++) {
    int n = n0 + t * 16 + lm;
    float bias = b3[n];
    #pragma unroll
    for (int r = 0; r < 8; r++) {
      int m = m0 + (half << 3) + r;
      float v = acc[t][r] + bias;
      H2[(size_t)m * FCH + n] = (v > 0.f) ? F8_P1 : 0;
    }
  }
}

__global__ void __launch_bounds__(256)
k_fc2(const uint8_t* __restrict__ H2, const uint8_t* __restrict__ W4b,
      const float* __restrict__ b4, float* __restrict__ out) {
  int lane = threadIdx.x & 31, wid = threadIdx.x >> 5;
  int half = lane >> 4, lm = lane & 15;
  int mt = blockIdx.x * 8 + wid;
  int m0 = mt << 4;
  const uint8_t* arow = H2 + (size_t)(m0 + lm) * FCH;
  const uint8_t* brow = W4b + (size_t)lm * FCH;

  v8f acc = {0.f, 0.f, 0.f, 0.f, 0.f, 0.f, 0.f, 0.f};
  #pragma unroll
  for (int kc = 0; kc < 8; kc++) {
    v16i a, bm;
    #pragma unroll
    for (int v = 0; v < 16; v++) {
      a[v]  = *(const int*)(arow + kc * 128 + kofsA(v, half));
      bm[v] = *(const int*)(brow + kc * 128 + kofsB(v, half));
    }
    acc = __builtin_amdgcn_wmma_f32_16x16x128_fp8_fp8(
              a, bm, (short)0, acc, false, false);
  }
  if (lm < 10) {
    float bias = b4[lm];
    #pragma unroll
    for (int r = 0; r < 8; r++) {
      int m = m0 + (half << 3) + r;
      out[m * 10 + lm] = acc[r] + bias;
    }
  }
}


extern "C" void kernel_launch(void* const* d_in, const int* in_sizes, int n_in,
                              void* d_out, int out_size, void* d_ws, size_t ws_size,
                              hipStream_t stream) {
  (void)in_sizes; (void)n_in; (void)out_size; (void)ws_size;
  const float* x  = (const float*)d_in[0];
  const float* w1 = (const float*)d_in[1];
  const float* b1 = (const float*)d_in[2];
  const float* w2 = (const float*)d_in[3];
  const float* b2 = (const float*)d_in[4];
  const float* w3 = (const float*)d_in[5];
  const float* b3 = (const float*)d_in[6];
  const float* w4 = (const float*)d_in[7];
  const float* b4 = (const float*)d_in[8];
  float* out = (float*)d_out;

  uint8_t* ws = (uint8_t*)d_ws;
  constexpr size_t OFF_W2B = 0;
  constexpr size_t OFF_W3B = OFF_W2B + (size_t)64 * K2P;
  constexpr size_t OFF_W4B = OFF_W3B + (size_t)FCH * K3P;
  constexpr size_t OFF_H1P = OFF_W4B + (size_t)16 * FCH;
  constexpr size_t OFF_C2B = OFF_H1P + (size_t)NB * H1P_IMG;
  constexpr size_t OFF_A1  = OFF_C2B + (size_t)NB * 196 * NC2;
  constexpr size_t OFF_H2  = OFF_A1  + (size_t)NB * K3P;
  uint8_t* W2b = ws + OFF_W2B;
  uint8_t* W3b = ws + OFF_W3B;
  uint8_t* W4b = ws + OFF_W4B;
  uint8_t* h1p = ws + OFF_H1P;
  uint8_t* c2b = ws + OFF_C2B;
  uint8_t* a1  = ws + OFF_A1;
  uint8_t* h2  = ws + OFF_H2;

  k_prep_w2<<<128,   256, 0, stream>>>(w2, W2b);
  k_prep_w3<<<10240, 256, 0, stream>>>(w3, W3b);
  k_prep_w4<<<64,    256, 0, stream>>>(w4, W4b);
  k_zero32 <<<10240, 256, 0, stream>>>((uint32_t*)a1, (int)((size_t)NB * K3P / 4));
  k_conv1  <<<5184,  256, 0, stream>>>(x, w1, b1, h1p);
  k_conv2  <<<6272,  256, 0, stream>>>(h1p, W2b, b2, c2b);
  k_pool2  <<<784,   256, 0, stream>>>(c2b, a1);
  k_fc1    <<<512,   256, 0, stream>>>(a1, W3b, b3, h2);
  k_fc2    <<<32,    256, 0, stream>>>(h2, W4b, b4, out);
}
